// VGAE_68917045231886
// MI455X (gfx1250) — matrix operand order repaired and checked by reading, not compiled, not run on hardware
//
#include <hip/hip_runtime.h>

// Problem constants (match reference)
#define NN 16384     // nodes
#define EE 524288    // edges
// F_IN=256, H1=64, H2=32

// ---------------------------------------------------------------------------
// Types for WMMA fragments (wave32, CDNA5 layouts per CDNA5 ISA)
// ---------------------------------------------------------------------------
typedef __attribute__((ext_vector_type(16))) __bf16        bf16x16;
typedef __attribute__((ext_vector_type(8)))  float         f32x8;
typedef __attribute__((ext_vector_type(4)))  float         f32x4;
typedef __attribute__((ext_vector_type(4)))  unsigned int  u32x4;

union BFrag {
  bf16x16        v;      // operand handed to the WMMA builtin (8 VGPRs)
  unsigned short u[16];  // element view (little-endian: u[0] = low half of VGPR0)
  u32x4          q[2];   // two 16-byte chunks for vector loads
};

__device__ __forceinline__ unsigned short f32_to_bf16_rne(float f) {
  unsigned int x = __float_as_uint(f);
  unsigned int r = 0x7FFFu + ((x >> 16) & 1u);   // round-to-nearest-even
  return (unsigned short)((x + r) >> 16);
}
__device__ __forceinline__ float bf16_bits_to_f32(unsigned short h) {
  return __uint_as_float(((unsigned int)h) << 16);
}
// fp32 -> (hi, lo) bf16 pair with v ~= hi + lo  (error ~2^-16 relative)
__device__ __forceinline__ void split_bf16(float v, unsigned short& hi, unsigned short& lo) {
  hi = f32_to_bf16_rne(v);
  lo = f32_to_bf16_rne(v - bf16_bits_to_f32(hi));
}

// C += Ah*Bh + Ah*Bl + Al*Bh  (drops Al*Bl, ~fp32 accuracy). 3x v_wmma_f32_16x16x32_bf16.
__device__ __forceinline__ f32x8 wmma3_bf16pair(const BFrag& ah, const BFrag& al,
                                                const BFrag& bh, const BFrag& bl,
                                                f32x8 acc) {
  acc = __builtin_amdgcn_wmma_f32_16x16x32_bf16(false, ah.v, false, bh.v, (short)0, acc, false, false);
  acc = __builtin_amdgcn_wmma_f32_16x16x32_bf16(false, ah.v, false, bl.v, (short)0, acc, false, false);
  acc = __builtin_amdgcn_wmma_f32_16x16x32_bf16(false, al.v, false, bh.v, (short)0, acc, false, false);
  return acc;
}

// ---------------------------------------------------------------------------
// Fragment loaders (build to ISA VGPR layouts, wave32)
//
// A 16x32 bf16: lane (m = lane&15, half = lane>>4) holds
//   elements 0..7  = A[m][kbase + 8*half + 0..7]
//   elements 8..15 = A[m][kbase + 16 + 8*half + 0..7]
// B 32x16 bf16: lane (n = lane&15, half) holds
//   elements 0..7  = B[kbase + 8*half + 0..7][n]
//   elements 8..15 = B[kbase + 16 + 8*half + 0..7][n]
// D 16x16 f32: VGPR r = C[r + 8*half][n = lane&15]
// ---------------------------------------------------------------------------
__device__ __forceinline__ void load_splitA_f32(const float* __restrict__ X, int ldx,
                                                int ibase, int kbase, int lane,
                                                BFrag& ah, BFrag& al) {
  const int m = lane & 15, half = lane >> 4;
  const float* p = X + (size_t)(ibase + m) * ldx + kbase;
  f32x4 c[4];
  c[0] = *(const f32x4*)(p + 8 * half);
  c[1] = *(const f32x4*)(p + 8 * half + 4);
  c[2] = *(const f32x4*)(p + 16 + 8 * half);
  c[3] = *(const f32x4*)(p + 16 + 8 * half + 4);
#pragma unroll
  for (int i = 0; i < 16; ++i) split_bf16(c[i >> 2][i & 3], ah.u[i], al.u[i]);
}

__device__ __forceinline__ void load_splitB_f32(const float* __restrict__ W, int ldw,
                                                int kbase, int jbase, int lane,
                                                BFrag& bh, BFrag& bl) {
  const int n = lane & 15, half = lane >> 4;
  const float* p = W + (size_t)(kbase + 8 * half) * ldw + jbase + n;
#pragma unroll
  for (int e = 0; e < 16; ++e) split_bf16(p[(size_t)(e + (e >> 3) * 8) * ldw], bh.u[e], bl.u[e]);
}

__device__ __forceinline__ void store_tile_f32(float* __restrict__ Y, int ldy,
                                               int ibase, int jbase, int lane,
                                               const f32x8& acc) {
  const int n = lane & 15, half = lane >> 4;
  float* p = Y + (size_t)(ibase + 8 * half) * ldy + jbase + n;
#pragma unroll
  for (int r = 0; r < 8; ++r) p[(size_t)r * ldy] = acc[r];
}

// ---------------------------------------------------------------------------
// K1: y1 = X[16384x256] @ W1[256x64]  (split-bf16 WMMA, f32 accumulate)
// grid 1024 blocks x 128 threads (4 waves; wave = column tile)
// ---------------------------------------------------------------------------
__global__ void __launch_bounds__(128) vgae_gemm_xw1(const float* __restrict__ X,
                                                     const float* __restrict__ W1,
                                                     float* __restrict__ Y1) {
  const int lane  = threadIdx.x & 31;
  const int wave  = threadIdx.x >> 5;       // 0..3
  const int ibase = blockIdx.x << 4;
  const int jbase = wave << 4;
  f32x8 acc = {0.f, 0.f, 0.f, 0.f, 0.f, 0.f, 0.f, 0.f};
#pragma unroll
  for (int kt = 0; kt < 8; ++kt) {          // K = 256 = 8 * 32
    BFrag ah, al, bh, bl;
    load_splitA_f32(X, 256, ibase, kt * 32, lane, ah, al);
    load_splitB_f32(W1, 64, kt * 32, jbase, lane, bh, bl);
    acc = wmma3_bf16pair(ah, al, bh, bl, acc);
  }
  store_tile_f32(Y1, 64, ibase, jbase, lane, acc);
}

// ---------------------------------------------------------------------------
// K2: h[src] += w_e * y1[dst]  (d = 64), one thread per (edge, feature)
// ---------------------------------------------------------------------------
__global__ void vgae_spmm64(const int* __restrict__ src, const int* __restrict__ dst,
                            const float* __restrict__ w, const float* __restrict__ hin,
                            float* __restrict__ hout) {
  const unsigned t = blockIdx.x * blockDim.x + threadIdx.x;
  const unsigned e = t >> 6, f = t & 63;
  const int s = src[e], d = dst[e];
  const float v = w[e] * hin[((size_t)d << 6) + f];
  __hip_atomic_fetch_add(&hout[((size_t)s << 6) + f], v,
                         __ATOMIC_RELAXED, __HIP_MEMORY_SCOPE_AGENT);
}

// ---------------------------------------------------------------------------
// K3: y2 = h @ W2, y3 = h @ W3  ([16384x64] @ [64x32], split-bf16 WMMA)
// grid 1024 x 128: waves 0,1 -> W2 cols 0-15/16-31; waves 2,3 -> W3
// ---------------------------------------------------------------------------
__global__ void __launch_bounds__(128) vgae_gemm_hw23(const float* __restrict__ H,
                                                      const float* __restrict__ W2,
                                                      const float* __restrict__ W3,
                                                      float* __restrict__ Y2,
                                                      float* __restrict__ Y3) {
  const int lane  = threadIdx.x & 31;
  const int wave  = threadIdx.x >> 5;       // 0..3 (uniform per wave)
  const float* W  = (wave < 2) ? W2 : W3;
  float*       Y  = (wave < 2) ? Y2 : Y3;
  const int jbase = (wave & 1) << 4;
  const int ibase = blockIdx.x << 4;
  f32x8 acc = {0.f, 0.f, 0.f, 0.f, 0.f, 0.f, 0.f, 0.f};
#pragma unroll
  for (int kt = 0; kt < 2; ++kt) {          // K = 64 = 2 * 32
    BFrag ah, al, bh, bl;
    load_splitA_f32(H, 64, ibase, kt * 32, lane, ah, al);
    load_splitB_f32(W, 32, kt * 32, jbase, lane, bh, bl);
    acc = wmma3_bf16pair(ah, al, bh, bl, acc);
  }
  store_tile_f32(Y, 32, ibase, jbase, lane, acc);
}

// ---------------------------------------------------------------------------
// K4: z_mean[src] += w*y2[dst], z_log_std[src] += w*y3[dst]  (d = 32)
// ---------------------------------------------------------------------------
__global__ void vgae_spmm32x2(const int* __restrict__ src, const int* __restrict__ dst,
                              const float* __restrict__ w,
                              const float* __restrict__ y2, const float* __restrict__ y3,
                              float* __restrict__ zm, float* __restrict__ zl) {
  const unsigned t = blockIdx.x * blockDim.x + threadIdx.x;
  const unsigned e = t >> 5, f = t & 31;
  const int s = src[e], d = dst[e];
  const float we = w[e];
  __hip_atomic_fetch_add(&zm[((size_t)s << 5) + f], we * y2[((size_t)d << 5) + f],
                         __ATOMIC_RELAXED, __HIP_MEMORY_SCOPE_AGENT);
  __hip_atomic_fetch_add(&zl[((size_t)s << 5) + f], we * y3[((size_t)d << 5) + f],
                         __ATOMIC_RELAXED, __HIP_MEMORY_SCOPE_AGENT);
}

// ---------------------------------------------------------------------------
// K5: zt = ((z * R) @ M) * R ; store as split-bf16 pair for the tensor GEMM
// ---------------------------------------------------------------------------
__global__ void vgae_dedicom_split(const float* __restrict__ zmean,
                                   const float* __restrict__ R, const float* __restrict__ M,
                                   unsigned short* __restrict__ zth,
                                   unsigned short* __restrict__ ztl) {
  const int t = blockIdx.x * blockDim.x + threadIdx.x;
  const int row = t >> 5, c = t & 31;
  const float* zr = zmean + ((size_t)row << 5);
  float s = 0.f;
#pragma unroll
  for (int k = 0; k < 32; ++k) s += zr[k] * R[k] * M[k * 32 + c];
  s *= R[c];
  unsigned short hi, lo;
  split_bf16(s, hi, lo);
  zth[t] = hi;
  ztl[t] = lo;
}

// ---------------------------------------------------------------------------
// K6: adj = zt @ zt.T  -- the big one. K=32 => ONE 16x16x32 WMMA pass per tile
// (x3 for the bf16-pair correction). zt hi/lo (2 MB total) is L2-resident;
// the 1.07 GB output streams out via nontemporal stores (TH=NT).
// grid 1024 blocks (tile rows) x 256 threads (8 waves, each owning j-tiles
// wave, wave+8, ..., 128 iterations).
// ---------------------------------------------------------------------------
__global__ void __launch_bounds__(256) vgae_adj_wmma(const unsigned short* __restrict__ zh,
                                                     const unsigned short* __restrict__ zl,
                                                     float* __restrict__ adj) {
  const int lane  = threadIdx.x & 31;
  const int wave  = threadIdx.x >> 5;
  const int ibase = blockIdx.x << 4;
  const int m = lane & 15, half = lane >> 4;

  // A fragments: rows ibase..ibase+15 of zt (loaded once, reused 128x)
  BFrag ah, al;
  {
    const unsigned short* pa = zh + ((size_t)(ibase + m) << 5);
    ah.q[0] = *(const u32x4*)(pa + 8 * half);
    ah.q[1] = *(const u32x4*)(pa + 16 + 8 * half);
    const unsigned short* pl = zl + ((size_t)(ibase + m) << 5);
    al.q[0] = *(const u32x4*)(pl + 8 * half);
    al.q[1] = *(const u32x4*)(pl + 16 + 8 * half);
  }

  for (int it = 0; it < 128; ++it) {
    const int jbase = ((it << 3) + wave) << 4;
    // B = zt.T tile: lane holds column n = m -> row (jbase+m) of zt, k-runs 8*half.. and 16+8*half..
    BFrag bh, bl;
    const unsigned short* pb = zh + ((size_t)(jbase + m) << 5) + 8 * half;
    bh.q[0] = *(const u32x4*)(pb);
    bh.q[1] = *(const u32x4*)(pb + 16);
    const unsigned short* pc = zl + ((size_t)(jbase + m) << 5) + 8 * half;
    bl.q[0] = *(const u32x4*)(pc);
    bl.q[1] = *(const u32x4*)(pc + 16);

    f32x8 acc = {0.f, 0.f, 0.f, 0.f, 0.f, 0.f, 0.f, 0.f};
    acc = wmma3_bf16pair(ah, al, bh, bl, acc);

    float* p = adj + ((size_t)(ibase + 8 * half) << 14) + jbase + m;  // row stride N=2^14
#pragma unroll
    for (int r = 0; r < 8; ++r)
      __builtin_nontemporal_store(acc[r], p + ((size_t)r << 14));
  }
}

// ---------------------------------------------------------------------------
// Host launcher
// d_out layout: adj[N*N] | z_mean[N*32] | z_log_std[N*32]  (fp32)
// Workspace (14 MB): y1(4M) | h(4M) | y2(2M) | y3(2M) | zt_hi(1M) | zt_lo(1M)
// ---------------------------------------------------------------------------
extern "C" void kernel_launch(void* const* d_in, const int* in_sizes, int n_in,
                              void* d_out, int out_size, void* d_ws, size_t ws_size,
                              hipStream_t stream) {
  (void)in_sizes; (void)n_in; (void)out_size; (void)ws_size;

  const float* x  = (const float*)d_in[0];
  const float* ew = (const float*)d_in[1];
  const float* W1 = (const float*)d_in[2];
  const float* W2 = (const float*)d_in[3];
  const float* W3 = (const float*)d_in[4];
  const float* R  = (const float*)d_in[5];
  const float* M  = (const float*)d_in[6];
  const int* esrc = (const int*)d_in[7];
  const int* edst = (const int*)d_in[8];

  float* adj   = (float*)d_out;
  float* zmean = adj + (size_t)NN * NN;
  float* zlog  = zmean + (size_t)NN * 32;

  char* ws = (char*)d_ws;
  float*          y1  = (float*)(ws);
  float*          h   = (float*)(ws + (size_t)4  * 1024 * 1024);
  float*          y2  = (float*)(ws + (size_t)8  * 1024 * 1024);
  float*          y3  = (float*)(ws + (size_t)10 * 1024 * 1024);
  unsigned short* zth = (unsigned short*)(ws + (size_t)12 * 1024 * 1024);
  unsigned short* ztl = (unsigned short*)(ws + (size_t)13 * 1024 * 1024);

  // Zero accumulation targets every launch (graph-capture safe).
  hipMemsetAsync(h, 0, (size_t)NN * 64 * sizeof(float), stream);
  hipMemsetAsync(zmean, 0, (size_t)NN * 64 * sizeof(float), stream);  // zmean+zlog

  vgae_gemm_xw1 <<<NN / 16, 128, 0, stream>>>(x, W1, y1);
  vgae_spmm64   <<<(EE * 64) / 256, 256, 0, stream>>>(esrc, edst, ew, y1, h);
  vgae_gemm_hw23<<<NN / 16, 128, 0, stream>>>(h, W2, W3, y2, y3);
  vgae_spmm32x2 <<<(EE * 32) / 256, 256, 0, stream>>>(esrc, edst, ew, y2, y3, zmean, zlog);
  vgae_dedicom_split<<<(NN * 32) / 256, 256, 0, stream>>>(zmean, R, M, zth, ztl);
  vgae_adj_wmma <<<NN / 16, 256, 0, stream>>>(zth, ztl, adj);
}
